// HeteroGNN_55130200211872
// MI455X (gfx1250) — compile-verified, matrix operand order checked by reading, not run on hardware
//
#include <hip/hip_runtime.h>

typedef __attribute__((ext_vector_type(16))) _Float16 v16h;
typedef __attribute__((ext_vector_type(8)))  _Float16 v8h;
typedef __attribute__((ext_vector_type(8)))  float    v8f;

#define NEG_SLOPE 0.2f

// ---------- helpers ----------
__device__ __forceinline__ unsigned f2ord(float f) {
    unsigned u = __float_as_uint(f);
    return (u & 0x80000000u) ? ~u : (u | 0x80000000u);
}
__device__ __forceinline__ float ord2f(unsigned u) {
    return (u & 0x80000000u) ? __uint_as_float(u & 0x7fffffffu)
                             : __uint_as_float(~u);
}
__device__ __forceinline__ float lrelu(float v) {
    return v > 0.f ? v : NEG_SLOPE * v;
}

// ---------- generic fills ----------
__global__ void fill_u32_k(unsigned* p, unsigned v, long n) {
    long i = (long)blockIdx.x * blockDim.x + threadIdx.x;
    if (i < n) p[i] = v;
}
__global__ void fill_f32_k(float* p, float v, long n) {
    long i = (long)blockIdx.x * blockDim.x + threadIdx.x;
    if (i < n) p[i] = v;
}

// ---------- f32 -> f16 convert ----------
__global__ void cvt_f16_k(const float* __restrict__ src, _Float16* __restrict__ dst, long n) {
    long i = (long)blockIdx.x * blockDim.x + threadIdx.x;
    if (i < n) dst[i] = (_Float16)src[i];
}

// ---------- layer-1 projection: [N,32]f16 @ [32,128]f16 -> [N,128]f16 via WMMA ----------
// grid.x = N/16 (M tiles), block = 256 (8 waves), wave w handles N-tile w (cols w*16..w*16+15).
// K = 32 exactly -> one v_wmma_f32_16x16x32_f16 per tile. No divergence: EXEC all ones.
__global__ void proj1_wmma_k(const _Float16* __restrict__ xh,
                             const _Float16* __restrict__ Wh,
                             _Float16* __restrict__ h1p) {
    __shared__ _Float16 ldsW[32 * 128];
    // cooperatively stage W (4096 halves = 2048 dwords) into LDS
    {
        const unsigned* gw = (const unsigned*)Wh;
        unsigned* lw = (unsigned*)ldsW;
        for (int i = threadIdx.x; i < 2048; i += 256) lw[i] = gw[i];
    }
    __syncthreads();

    const int lane  = threadIdx.x & 31;
    const int wave  = threadIdx.x >> 5;     // 0..7 -> N tile
    const int nbase = wave * 16;

    // ---- A fragment: 16x32 f16. lane L<16: row=L, K={0..7,16..23}; lane L+16: K={8..15,24..31}
    const int  r   = lane & 15;
    const int  kh  = lane >> 4;
    const long row = (long)blockIdx.x * 16 + r;
    const _Float16* xr = xh + row * 32;
    v8h a0 = *(const v8h*)(xr + kh * 8);
    v8h a1 = *(const v8h*)(xr + 16 + kh * 8);
    v16h a;
#pragma unroll
    for (int i = 0; i < 8; ++i) { a[i] = a0[i]; a[8 + i] = a1[i]; }

    // ---- B fragment: 32x16 f16, lane n<16: col=nbase+n, K={0..7,16..23}; lane n+16: K={8..15,24..31}
    const int n   = lane & 15;
    const int col = nbase + n;
    v16h b;
#pragma unroll
    for (int i = 0; i < 8; ++i) {
        b[i]     = ldsW[(kh * 8 + i) * 128 + col];
        b[8 + i] = ldsW[(16 + kh * 8 + i) * 128 + col];
    }

    v8f c = {};
    c = __builtin_amdgcn_wmma_f32_16x16x32_f16(false, a, false, b, (short)0, c, false, false);

    // ---- D store: lane L: col=L&15, rows = (L>>4)*8 + r, VGPR r holds row r (+8 for hi half)
    const int coln = lane & 15;
    const int rb   = (lane >> 4) * 8;
#pragma unroll
    for (int rr = 0; rr < 8; ++rr) {
        long orow = (long)blockIdx.x * 16 + rb + rr;
        h1p[orow * 128 + nbase + coln] = (_Float16)c[rr];
    }
}

// ---------- per-node attention logits, layer 1: el/er [N,2] ----------
__global__ void elr1_k(const _Float16* __restrict__ h1p,
                       const float* __restrict__ aL, const float* __restrict__ aR,
                       float* __restrict__ el, float* __restrict__ er, long n2) {
    long idx = (long)blockIdx.x * blockDim.x + threadIdx.x;
    if (idx >= n2) return;
    long nid = idx >> 1;
    int  h   = (int)(idx & 1);
    float sl = 0.f, sr = 0.f;
    const _Float16* hp = h1p + nid * 128 + h * 64;
#pragma unroll 4
    for (int o = 0; o < 64; ++o) {
        float v = (float)hp[o];
        sl += v * aL[h * 64 + o];
        sr += v * aR[h * 64 + o];
    }
    el[idx] = sl;
    er[idx] = sr;
}

// ---------- edge pass A: atomic max of edge scores per (dst, head) ----------
__global__ void edge_max_k(const int* __restrict__ src, const int* __restrict__ dst,
                           const float* __restrict__ el, const float* __restrict__ er,
                           unsigned* __restrict__ mEnc, int E) {
    int e = blockIdx.x * blockDim.x + threadIdx.x;
    if (e >= E) return;
    int s = src[e], d = dst[e];
#pragma unroll
    for (int h = 0; h < 2; ++h) {
        float v = lrelu(el[s * 2 + h] + er[d * 2 + h]);
        atomicMax(&mEnc[d * 2 + h], f2ord(v));
    }
}

// ---------- edge pass B (layer 1): one wave per edge; lanes split the 128 features ----------
__global__ void edge_acc1_k(const int* __restrict__ src, const int* __restrict__ dst,
                            const float* __restrict__ el, const float* __restrict__ er,
                            const unsigned* __restrict__ mEnc, float* __restrict__ dsum,
                            const _Float16* __restrict__ h1p, float* __restrict__ h1cat,
                            int relOff, int E) {
    long gid  = (long)blockIdx.x * blockDim.x + threadIdx.x;
    long wave = gid >> 5;
    int  lane = (int)(gid & 31);
    if (wave >= E) return;
    int s = src[wave], d = dst[wave];

    float ee0 = __expf(lrelu(el[s * 2 + 0] + er[d * 2 + 0]) - ord2f(mEnc[d * 2 + 0]));
    float ee1 = __expf(lrelu(el[s * 2 + 1] + er[d * 2 + 1]) - ord2f(mEnc[d * 2 + 1]));
    if (lane == 0) {
        atomicAdd(&dsum[d * 2 + 0], ee0);
        atomicAdd(&dsum[d * 2 + 1], ee1);
    }
    const _Float16* hp = h1p + (long)s * 128;
    float* op = h1cat + (long)d * 256 + relOff;
#pragma unroll
    for (int j = 0; j < 4; ++j) {
        int   c  = lane * 4 + j;           // 0..127
        float ee = (c < 64) ? ee0 : ee1;
        atomicAdd(&op[c], ee * (float)hp[c]);
    }
}

// ---------- pass C (layer 1): normalize + bias, in place in h1cat ----------
__global__ void norm1_k(float* __restrict__ h1cat, const float* __restrict__ dsum,
                        const float* __restrict__ b1, int relOff, long n128) {
    long idx = (long)blockIdx.x * blockDim.x + threadIdx.x;
    if (idx >= n128) return;
    long nid = idx >> 7;
    int  c   = (int)(idx & 127);
    float dn = dsum[nid * 2 + (c >> 6)];
    float* p = h1cat + nid * 256 + relOff + c;
    float  s = *p;
    *p = (dn > 0.f ? s / dn : 0.f) + b1[c];
}

// ---------- layer-2 projection + logits: h2p[n,h] = h1cat[n,:] . W2[:,h] ----------
__global__ void proj2_k(const float* __restrict__ h1cat, const float* __restrict__ W2,
                        const float* __restrict__ aL2, const float* __restrict__ aR2,
                        float* __restrict__ h2p, float* __restrict__ el2,
                        float* __restrict__ er2, long n2) {
    long idx = (long)blockIdx.x * blockDim.x + threadIdx.x;
    if (idx >= n2) return;
    long nid = idx >> 1;
    int  h   = (int)(idx & 1);
    const float* row = h1cat + nid * 256;
    float s = 0.f;
#pragma unroll 8
    for (int k = 0; k < 256; ++k) s += row[k] * W2[k * 2 + h];
    h2p[idx] = s;
    el2[idx] = s * aL2[h];
    er2[idx] = s * aR2[h];
}

// ---------- layer-2 edge passes (O=1 per head) ----------
__global__ void edge_acc2_k(const int* __restrict__ src, const int* __restrict__ dst,
                            const float* __restrict__ el, const float* __restrict__ er,
                            const unsigned* __restrict__ mEnc, float* __restrict__ dsum,
                            const float* __restrict__ h2p, float* __restrict__ out,
                            int relOff, int E) {
    int e = blockIdx.x * blockDim.x + threadIdx.x;
    if (e >= E) return;
    int s = src[e], d = dst[e];
#pragma unroll
    for (int h = 0; h < 2; ++h) {
        float ee = __expf(lrelu(el[s * 2 + h] + er[d * 2 + h]) - ord2f(mEnc[d * 2 + h]));
        atomicAdd(&dsum[d * 2 + h], ee);
        atomicAdd(&out[d * 4 + relOff + h], ee * h2p[s * 2 + h]);
    }
}

__global__ void norm2_k(float* __restrict__ out, const float* __restrict__ dsum,
                        const float* __restrict__ b2, int relOff, long n2) {
    long idx = (long)blockIdx.x * blockDim.x + threadIdx.x;
    if (idx >= n2) return;
    long nid = idx >> 1;
    int  h   = (int)(idx & 1);
    float dn = dsum[idx];
    float* p = out + nid * 4 + relOff + h;
    *p = (dn > 0.f ? (*p) / dn : 0.f) + b2[h];
}

// ==================== launch ====================
extern "C" void kernel_launch(void* const* d_in, const int* in_sizes, int n_in,
                              void* d_out, int out_size, void* d_ws, size_t ws_size,
                              hipStream_t stream) {
    const float* x   = (const float*)d_in[0];
    const float* W1  = (const float*)d_in[1];
    const float* aL1 = (const float*)d_in[2];
    const float* aR1 = (const float*)d_in[3];
    const float* b1  = (const float*)d_in[4];
    const float* W2  = (const float*)d_in[5];
    const float* aL2 = (const float*)d_in[6];
    const float* aR2 = (const float*)d_in[7];
    const float* b2  = (const float*)d_in[8];
    const int* srcs[2] = {(const int*)d_in[9],  (const int*)d_in[11]};
    const int* dsts[2] = {(const int*)d_in[10], (const int*)d_in[12]};

    const long Nn = in_sizes[0] / 32;   // 50000 (multiple of 16)
    const int  E  = in_sizes[9];        // 800000
    float* out = (float*)d_out;

    // workspace carve-out (256B aligned)
    char* base = (char*)d_ws;
    size_t off = 0;
    auto carve = [&](size_t bytes) -> char* {
        char* p = base + off;
        off = (off + bytes + 255) & ~(size_t)255;
        return p;
    };
    _Float16* xh    = (_Float16*)carve(Nn * 32 * 2);
    _Float16* W1h   = (_Float16*)carve(32 * 128 * 2);
    _Float16* h1p   = (_Float16*)carve(Nn * 128 * 2);
    float*    el1   = (float*)carve(Nn * 2 * 4);
    float*    er1   = (float*)carve(Nn * 2 * 4);
    unsigned* m1    = (unsigned*)carve(Nn * 2 * 4);
    float*    d1    = (float*)carve(Nn * 2 * 4);
    float*    h1cat = (float*)carve(Nn * 256 * 4);
    float*    h2p   = (float*)carve(Nn * 2 * 4);
    float*    el2   = (float*)carve(Nn * 2 * 4);
    float*    er2   = (float*)carve(Nn * 2 * 4);
    unsigned* m2    = (unsigned*)carve(Nn * 2 * 4);
    float*    d2    = (float*)carve(Nn * 2 * 4);
    (void)ws_size;

    const unsigned NEG_INF_ORD = 0x007FFFFFu;  // f2ord(-inf)
    auto blocks = [](long n, int bs) { return (unsigned)((n + bs - 1) / bs); };

    // 1. convert inputs to f16
    cvt_f16_k<<<blocks(Nn * 32, 256), 256, 0, stream>>>(x, xh, Nn * 32);
    cvt_f16_k<<<blocks(32 * 128, 256), 256, 0, stream>>>(W1, W1h, 32 * 128);

    // 2. layer-1 projection via WMMA (3125 M-tiles x 8 waves covering 128 cols)
    proj1_wmma_k<<<(unsigned)(Nn / 16), 256, 0, stream>>>(xh, W1h, h1p);

    // 3. attention logits
    elr1_k<<<blocks(Nn * 2, 256), 256, 0, stream>>>(h1p, aL1, aR1, el1, er1, Nn * 2);

    // 4. zero concat accumulator once (both relations write disjoint column halves)
    fill_f32_k<<<blocks(Nn * 256, 256), 256, 0, stream>>>(h1cat, 0.f, Nn * 256);

    for (int rel = 0; rel < 2; ++rel) {
        const int relOff = rel * 128;
        fill_u32_k<<<blocks(Nn * 2, 256), 256, 0, stream>>>(m1, NEG_INF_ORD, Nn * 2);
        fill_f32_k<<<blocks(Nn * 2, 256), 256, 0, stream>>>(d1, 0.f, Nn * 2);
        edge_max_k<<<blocks(E, 256), 256, 0, stream>>>(srcs[rel], dsts[rel], el1, er1, m1, E);
        edge_acc1_k<<<blocks((long)E * 32, 256), 256, 0, stream>>>(
            srcs[rel], dsts[rel], el1, er1, m1, d1, h1p, h1cat, relOff, E);
        norm1_k<<<blocks(Nn * 128, 256), 256, 0, stream>>>(h1cat, d1, b1, relOff, Nn * 128);
    }

    // 5. layer-2 projection + logits
    proj2_k<<<blocks(Nn * 2, 256), 256, 0, stream>>>(h1cat, W2, aL2, aR2, h2p, el2, er2, Nn * 2);

    // 6. zero output accumulator, then layer-2 edge passes per relation
    fill_f32_k<<<blocks(Nn * 4, 256), 256, 0, stream>>>(out, 0.f, Nn * 4);
    for (int rel = 0; rel < 2; ++rel) {
        const int relOff = rel * 2;
        fill_u32_k<<<blocks(Nn * 2, 256), 256, 0, stream>>>(m2, NEG_INF_ORD, Nn * 2);
        fill_f32_k<<<blocks(Nn * 2, 256), 256, 0, stream>>>(d2, 0.f, Nn * 2);
        edge_max_k<<<blocks(E, 256), 256, 0, stream>>>(srcs[rel], dsts[rel], el2, er2, m2, E);
        edge_acc2_k<<<blocks(E, 256), 256, 0, stream>>>(
            srcs[rel], dsts[rel], el2, er2, m2, d2, h2p, out, relOff, E);
        norm2_k<<<blocks(Nn * 2, 256), 256, 0, stream>>>(out, d2, b2, relOff, Nn * 2);
    }
}
